// SimpleDiscriminator_28836410425363
// MI455X (gfx1250) — compile-verified, matrix operand order checked by reading, not run on hardware
//
#include <hip/hip_runtime.h>
#include <math.h>

typedef __attribute__((ext_vector_type(2)))  float    v2f;
typedef __attribute__((ext_vector_type(8)))  float    v8f;
typedef __attribute__((ext_vector_type(16))) _Float16 v16h;
typedef __attribute__((ext_vector_type(4)))  unsigned int u32x4;
typedef __attribute__((ext_vector_type(8)))  int      i32x8;
typedef __attribute__((ext_vector_type(4)))  int      i32x4;

#define FIN 512   // node feature width (fixed by reference)

// ---------------------------------------------------------------------------
// 1) deg[i] = 1.0 (self loop weight), zero global reduction scalar
// ---------------------------------------------------------------------------
__global__ void gcn_init_kernel(float* __restrict__ deg, float* __restrict__ red, int n) {
    int i = blockIdx.x * blockDim.x + threadIdx.x;
    if (i < n) deg[i] = 1.0f;
    if (i == 0) *red = 0.0f;
}

// ---------------------------------------------------------------------------
// 2) deg[col[e]] += edge_attr[e]
// ---------------------------------------------------------------------------
__global__ void gcn_deg_kernel(const long long* __restrict__ el,
                               const float* __restrict__ ea,
                               float* __restrict__ deg, long long E) {
    long long e = (long long)blockIdx.x * blockDim.x + threadIdx.x;
    if (e < E) {
        int c = (int)el[E + e];
        atomicAdd(&deg[c], ea[e]);
    }
}

// ---------------------------------------------------------------------------
// 3) deg -> deg^(-1/2)  (in place; deg >= 1 because of self loops)
// ---------------------------------------------------------------------------
__global__ void gcn_rsqrt_kernel(float* __restrict__ deg, int n) {
    int i = blockIdx.x * blockDim.x + threadIdx.x;
    if (i < n) {
        float d = deg[i];
        deg[i] = (d > 0.0f) ? rsqrtf(d) : 0.0f;
    }
}

// ---------------------------------------------------------------------------
// 4) WMMA GEMM: h = x @ W  (one wave32 per 16-row tile, K=512).
//    W (4KB) staged in LDS via the Tensor Data Mover (TDM) when available.
//    Seeds acc with the self-loop term: acc = h * dis^2.
// ---------------------------------------------------------------------------
__global__ void gcn_gemm_wmma_kernel(const float* __restrict__ x,
                                     const float* __restrict__ W,
                                     const float* __restrict__ dis,
                                     float* __restrict__ h,
                                     float* __restrict__ acc, int n) {
    __shared__ __align__(16) float ldsW[FIN * 2];   // row-major copy of W [512,2]

    const int lane = threadIdx.x & 31;
    const int l    = lane & 15;   // row (A) / column (B,C) within half-wave
    const int hi   = lane >> 4;   // half-wave select
    const int ntiles = (n + 15) >> 4;
    const int tile = blockIdx.x * (blockDim.x >> 5) + (threadIdx.x >> 5);
    const bool valid = (tile < ntiles);

    // ---- stage W into LDS --------------------------------------------------
#if __has_builtin(__builtin_amdgcn_tensor_load_to_lds) && \
    __has_builtin(__builtin_amdgcn_s_wait_tensorcnt)
    if ((threadIdx.x >> 5) == 0) {          // wave 0 issues one TDM transfer
        unsigned long long ga = (unsigned long long)(const void*)W;
        unsigned int la = (unsigned int)(unsigned long long)(const void*)&ldsW[0];
        // D# group 0: count=1 | lds_addr | global_addr[56:0] | type=2
        u32x4 g0;
        g0[0] = 1u;                                        // count=1, user mode
        g0[1] = la;                                        // LDS byte address
        g0[2] = (unsigned int)(ga & 0xFFFFFFFFu);          // global_addr[31:0]
        g0[3] = (unsigned int)((ga >> 32) & 0x01FFFFFFu)   // global_addr[56:32]
              | 0x80000000u;                               // type=2 ("image")
        // D# group 1: data_size=4B, tensor 1024x1, tile 1024x1, stride 1024
        i32x8 g1;
        g1[0] = (2 << 16);                 // workgroup_mask=0, data_size=2 (4B)
        g1[1] = (1024 << 16);              // tensor_dim0[15:0]=1024
        g1[2] = (int)((1024u >> 16)        // tensor_dim0[31:16] = 0
              | (1u << 16));               // tensor_dim1[15:0] = 1
        g1[3] = (1024 << 16);              // tensor_dim1[31:16]=0 | tile_dim0=1024
        g1[4] = 1;                         // tile_dim1=1, tile_dim2=0
        g1[5] = 1024;                      // tensor_dim0_stride[31:0]
        g1[6] = (1024 << 16);              // stride[47:32]=0 | tensor_dim1_stride[15:0]
        g1[7] = 0;                         // tensor_dim1_stride[47:16]=0
        i32x4 gz  = {0, 0, 0, 0};          // groups 2/3 unused (2D tensor)
        i32x8 gz8 = {0, 0, 0, 0, 0, 0, 0, 0};
        __builtin_amdgcn_tensor_load_to_lds(g0, g1, gz, gz, gz8, 0 /*cpol*/);
        __builtin_amdgcn_s_wait_tensorcnt(0);
    }
#else
    for (int i = threadIdx.x; i < FIN * 2; i += blockDim.x) ldsW[i] = W[i];
#endif
    __syncthreads();

    if (!valid) return;                     // wave-uniform; after barrier

    const int  row0 = tile * 16;
    const int  rowL = (row0 + l < n) ? (row0 + l) : (n - 1);
    const float* __restrict__ xrow = x + (long long)rowL * FIN;

    // lane masks: only B columns 0,1 are non-zero (W has F_OUT=2)
    const float m0 = (l == 0) ? 1.0f : 0.0f;
    const float m1 = (l == 1) ? 1.0f : 0.0f;

    v8f c = {};
#if __has_builtin(__builtin_amdgcn_wmma_f32_16x16x4_f32)
    // A 16x4 f32: lane(l,hi) holds K = k+2*hi, k+2*hi+1 for row M=l
    // B 4x16 f32: lane(l,hi) holds K = k+2*hi, k+2*hi+1 for col N=l
    for (int k = 0; k < FIN; k += 4) {
        float2 av = *(const float2*)(xrow + k + 2 * hi);
        float4 q  = *(const float4*)(&ldsW[(k + 2 * hi) * 2]);  // broadcast ds_load_b128
        v2f a, b;
        a[0] = av.x;
        a[1] = av.y;
        b[0] = m0 * q.x + m1 * q.y;   // W[k+2hi  ][l] for l<2, else 0
        b[1] = m0 * q.z + m1 * q.w;   // W[k+2hi+1][l] for l<2, else 0
        c = __builtin_amdgcn_wmma_f32_16x16x4_f32(false, a, false, b,
                                                  (short)0, c, false, false);
    }
#else
    for (int k = 0; k < FIN; k += 32) {
        v16h a, b;
#pragma unroll
        for (int j = 0; j < 16; ++j) {
            int kk = (j < 8) ? (j + 8 * hi) : (j - 8 + 16 + 8 * hi);
            float w0 = ldsW[(k + kk) * 2 + 0];
            float w1 = ldsW[(k + kk) * 2 + 1];
            a[j] = (_Float16)xrow[k + kk];
            b[j] = (_Float16)(m0 * w0 + m1 * w1);
        }
        c = __builtin_amdgcn_wmma_f32_16x16x32_f16(false, a, false, b,
                                                   (short)0, c, false, false);
    }
#endif
    // C/D 16x16 f32 layout: VGPR v, lane(l,hi) -> element [M = v + 8*hi][N = l]
    if (l < 2) {
#pragma unroll
        for (int v = 0; v < 8; ++v) {
            int node = row0 + v + 8 * hi;
            if (node < n) {
                float val = c[v];
                float ds  = dis[node];
                h[node * 2 + l]   = val;
                acc[node * 2 + l] = val * ds * ds;   // self-loop norm = dis^2
            }
        }
    }
}

// ---------------------------------------------------------------------------
// 5) acc[col] += h[row] * (dis[row] * w * dis[col])
// ---------------------------------------------------------------------------
__global__ void gcn_edge_kernel(const long long* __restrict__ el,
                                const float* __restrict__ ea,
                                const float* __restrict__ dis,
                                const float* __restrict__ h,
                                float* __restrict__ acc, long long E) {
    long long e = (long long)blockIdx.x * blockDim.x + threadIdx.x;
    if (e < E) {
        __builtin_prefetch(&el[e + 4096], 0, 0);   // global_prefetch_b8
        int   r = (int)el[e];
        int   c = (int)el[E + e];
        float s = dis[r] * ea[e] * dis[c];
        atomicAdd(&acc[c * 2 + 0], h[r * 2 + 0] * s);
        atomicAdd(&acc[c * 2 + 1], h[r * 2 + 1] * s);
    }
}

// ---------------------------------------------------------------------------
// 6) red += sum_i relu(acc[i] + b[i&1]) * fc_w[i]
// ---------------------------------------------------------------------------
__global__ void gcn_reduce_kernel(const float* __restrict__ acc,
                                  const float* __restrict__ b,
                                  const float* __restrict__ fcw,
                                  float* __restrict__ red, int total) {
    __shared__ float sm[256];
    float sum = 0.0f;
    for (int i = blockIdx.x * blockDim.x + threadIdx.x; i < total;
         i += gridDim.x * blockDim.x) {
        float v = acc[i] + b[i & 1];
        v = (v > 0.0f) ? v : 0.0f;
        sum += v * fcw[i];
    }
    sm[threadIdx.x] = sum;
    __syncthreads();
    for (int s = blockDim.x >> 1; s > 0; s >>= 1) {
        if (threadIdx.x < s) sm[threadIdx.x] += sm[threadIdx.x + s];
        __syncthreads();
    }
    if (threadIdx.x == 0) atomicAdd(red, sm[0]);
}

// ---------------------------------------------------------------------------
// 7) out = sigmoid(red + fc_b)
// ---------------------------------------------------------------------------
__global__ void gcn_final_kernel(const float* __restrict__ red,
                                 const float* __restrict__ fcb,
                                 float* __restrict__ out) {
    if (blockIdx.x == 0 && threadIdx.x == 0) {
        float z = *red + *fcb;
        out[0] = 1.0f / (1.0f + expf(-z));
    }
}

// ---------------------------------------------------------------------------
extern "C" void kernel_launch(void* const* d_in, const int* in_sizes, int n_in,
                              void* d_out, int out_size, void* d_ws, size_t ws_size,
                              hipStream_t stream) {
    const float*     x   = (const float*)d_in[0];      // [N, 512] f32
    const long long* el  = (const long long*)d_in[1];  // [2, E]   i64
    const float*     ea  = (const float*)d_in[2];      // [E]      f32
    const float*     W   = (const float*)d_in[3];      // [512, 2] f32
    const float*     b   = (const float*)d_in[4];      // [2]      f32
    const float*     fcw = (const float*)d_in[5];      // [2N]     f32
    const float*     fcb = (const float*)d_in[6];      // scalar   f32
    float* out = (float*)d_out;

    const int       N = in_sizes[5] / 2;
    const long long E = (long long)(in_sizes[1] / 2);

    // workspace layout (floats): deg/dis [N] | h [2N] | acc [2N] | red [1]
    float* deg = (float*)d_ws;
    float* h   = deg + N;
    float* acc = h + 2 * (long long)N;
    float* red = acc + 2 * (long long)N;

    const int T = 256;
    const int nBlk = (N + T - 1) / T;
    const int eBlk = (int)((E + T - 1) / T);

    gcn_init_kernel <<<nBlk, T, 0, stream>>>(deg, red, N);
    gcn_deg_kernel  <<<eBlk, T, 0, stream>>>(el, ea, deg, E);
    gcn_rsqrt_kernel<<<nBlk, T, 0, stream>>>(deg, N);

    const int ntiles = (N + 15) / 16;      // one wave32 per 16-row tile
    const int wavesPerBlock = 8;           // 256 threads, 8 tiles share W staging
    gcn_gemm_wmma_kernel<<<(ntiles + wavesPerBlock - 1) / wavesPerBlock,
                           wavesPerBlock * 32, 0, stream>>>(x, W, deg, h, acc, N);

    gcn_edge_kernel <<<eBlk, T, 0, stream>>>(el, ea, deg, h, acc, E);
    gcn_reduce_kernel<<<1024, T, 0, stream>>>(acc, b, fcw, red, 2 * N);
    gcn_final_kernel<<<1, 32, 0, stream>>>(red, fcb, out);
}
